// StereoFusionModule_49615462203400
// MI455X (gfx1250) — compile-verified, matrix operand order checked by reading, not run on hardware
//
#include <hip/hip_runtime.h>
#include <hip/hip_bf16.h>
#include <stdint.h>

typedef __attribute__((ext_vector_type(16))) _Float16 v16h;
typedef __attribute__((ext_vector_type(8)))  _Float16 v8h;
typedef __attribute__((ext_vector_type(4)))  _Float16 v4h;
typedef __attribute__((ext_vector_type(8)))  float    v8f;

#define DEV __device__ __forceinline__

DEV v8f wmma32(v16h a, v16h b, v8f c) {
  // v_wmma_f32_16x16x32_f16 : D = A(16x32 f16) * B(32x16 f16) + C(16x16 f32)
  return __builtin_amdgcn_wmma_f32_16x16x32_f16(false, a, false, b, (short)0, c,
                                                false, false);
}

// Fragment loader for the 16-bit A layout (and B via a [n][k] transposed tile):
// lane l holds m = l&15; first 8 halves are k = (l>>4)*8 + 0..7, next 8 are
// k = 16 + (l>>4)*8 + 0..7 (CDNA5 ISA, 16-bit A-matrix 16x32).
// Works on LDS or global pointers; rows must be 16B-aligned (ld multiple of 8).
DEV v16h ld_frag(const _Float16* __restrict__ p0, int ld, int r0, int k0) {
  const int lane = threadIdx.x & 31;
  const _Float16* p = p0 + (size_t)(r0 + (lane & 15)) * ld + k0 + (lane >> 4) * 8;
  union { v16h v; v8h h[2]; } u;
  u.h[0] = *(const v8h*)p;
  u.h[1] = *(const v8h*)(p + 16);
  return u.v;
}

// ---- ASYNC copy helpers: GLOBAL -> LDS, tracked by ASYNCcnt -----------------
// Per-lane semantics (CDNA5 ISA $4.4): each EXEC lane moves 16B
// from its global address to its LDS address. LDS offset = low 32 bits of the
// flat shared-pointer address.

// 128 x 32 halves tile, LDS row stride 40 halves (80B, 16B-aligned rows).
DEV void async_tile_A(uint32_t lds_base, const _Float16* __restrict__ g,
                      int ldg, int tid) {
#pragma unroll
  for (int rnd = 0; rnd < 2; ++rnd) {          // 2 async instructions / wave
    int e = (rnd * 256 + tid) * 8;
    int r = e >> 5, c = e & 31;
    uint32_t lds = lds_base + (uint32_t)(r * 40 + c) * 2;
    const _Float16* src = g + (size_t)r * ldg + c;
    asm volatile("global_load_async_to_lds_b128 %0, %1, off"
                 :: "v"(lds), "v"(src) : "memory");
  }
}

// 128 x 128 halves tile, LDS row stride 136 halves (272B, 16B-aligned rows).
DEV void async_tile_128(uint32_t lds_base, const _Float16* __restrict__ g,
                        int ldg, int tid) {
#pragma unroll
  for (int rnd = 0; rnd < 8; ++rnd) {          // 8 async instructions / wave
    int e = (rnd * 256 + tid) * 8;
    int r = e >> 7, c = e & 127;
    uint32_t lds = lds_base + (uint32_t)(r * 136 + c) * 2;
    const _Float16* src = g + (size_t)r * ldg + c;
    asm volatile("global_load_async_to_lds_b128 %0, %1, off"
                 :: "v"(lds), "v"(src) : "memory");
  }
}

// ---------------------------------------------------------------------------
// Generic 128x128-tile GEMM, 256 threads = 8 waves, each wave owns 16 rows x
// 128 cols (8 accumulator tiles). BK = 32 (one WMMA K-depth per stage).
// MODE 0: scale GEMM  (A fp32 concat layout, +bias, write fp32 + f16)
// MODE 1: Q/K proj    (A f16 async-staged, +bias, fused per-head RMSNorm)
// MODE 2: V proj      (A f16 async-staged, +bias, write V^T (B,H,Dh,N) f16)
// MODE 3: out proj    (A f16 async-staged, +bias, +residual, fp32 pre-LN)
// A tiles for f16 modes are double-buffered via GLOBAL_LOAD_ASYNC_TO_LDS.
// ---------------------------------------------------------------------------
template <int MODE>
__global__ void __launch_bounds__(256)
gemm_k(const float* __restrict__ Af32, const _Float16* __restrict__ Af16,
       const float* __restrict__ W, const float* __restrict__ bias,
       const float* __restrict__ nrmw, const float* __restrict__ resid,
       float* __restrict__ outF, _Float16* __restrict__ out16, int Kdim)
{
  __shared__ __align__(16) _Float16 As[2][128][40];  // [m][k], 80B rows
  __shared__ __align__(16) _Float16 Bs[128][40];     // [n][k] (transposed W)

  const int tid  = threadIdx.x;
  const int w    = tid >> 5;
  const int lane = tid & 31;
  const int hi   = lane >> 4;
  const int ln   = lane & 15;
  const int n0   = blockIdx.x * 128;
  const int m0   = blockIdx.y * 128;

  v8f acc[8];
#pragma unroll
  for (int t = 0; t < 8; ++t) acc[t] = 0.f;

  const int ksteps = Kdim >> 5;
  if (MODE != 0)   // prefetch first A tile asynchronously into buffer 0
    async_tile_A((uint32_t)(uintptr_t)&As[0][0][0],
                 Af16 + (size_t)m0 * Kdim, Kdim, tid);

  for (int kt = 0; kt < ksteps; ++kt) {
    const int k0  = kt << 5;
    const int buf = kt & 1;
    __syncthreads();   // previous compute done reading Bs / As buffers

    // ---- B tile (32 x 128) -> LDS transposed [n][k], fp32 -> f16
    {
      const float* baseB = W + (size_t)k0 * 1024 + n0;
#pragma unroll
      for (int rnd = 0; rnd < 4; ++rnd) {
        int e = (rnd * 256 + tid) * 4;
        int kr = e >> 7, nc = e & 127;
        float4 f = *(const float4*)(baseB + (size_t)kr * 1024 + nc);
        Bs[nc + 0][kr] = (_Float16)f.x;
        Bs[nc + 1][kr] = (_Float16)f.y;
        Bs[nc + 2][kr] = (_Float16)f.z;
        Bs[nc + 3][kr] = (_Float16)f.w;
      }
    }

    if (MODE == 0) {
      // ---- A tile (128 x 32) fp32 concat layout -> LDS f16 (manual convert)
      const int b = m0 >> 10, nbase = m0 & 1023;
      const int s = k0 >> 10, d0 = k0 & 1023;
      const float* baseA =
          Af32 + (((size_t)b * 3 + s) * 1024 + nbase) * 1024 + d0;
#pragma unroll
      for (int rnd = 0; rnd < 4; ++rnd) {
        int e = (rnd * 256 + tid) * 4;
        int r = e >> 5, c = e & 31;
        float4 f = *(const float4*)(baseA + (size_t)r * 1024 + c);
        v4h hv;
        hv[0] = (_Float16)f.x; hv[1] = (_Float16)f.y;
        hv[2] = (_Float16)f.z; hv[3] = (_Float16)f.w;
        *(v4h*)&As[buf][r][c] = hv;
      }
    } else {
      // ---- issue async prefetch of tile kt+1, retire tile kt (in-order)
      if (kt + 1 < ksteps) {
        async_tile_A((uint32_t)(uintptr_t)&As[(kt + 1) & 1][0][0],
                     Af16 + (size_t)m0 * Kdim + (k0 + 32), Kdim, tid);
        asm volatile("s_wait_asynccnt 0x2" ::: "memory");
      } else {
        asm volatile("s_wait_asynccnt 0x0" ::: "memory");
      }
    }

    // hint next W tile into cache while we compute
    if (kt + 1 < ksteps)
      __builtin_prefetch(W + (size_t)(k0 + 32 + (tid >> 3)) * 1024 + n0 +
                             (tid & 7) * 16, 0, 1);

    __syncthreads();   // tiles visible to all waves

    v16h a = ld_frag(&As[buf][0][0], 40, w * 16, 0);
#pragma unroll
    for (int t = 0; t < 8; ++t) {
      v16h b = ld_frag(&Bs[0][0], 40, t * 16, 0);
      acc[t] = wmma32(a, b, acc[t]);
    }
  }

  // ---- epilogue, fully in registers (rows live inside one wave)
  const int rowbase = m0 + w * 16 + hi * 8;
  if (MODE == 0) {
#pragma unroll
    for (int r = 0; r < 8; ++r) {
      const size_t row = (size_t)(rowbase + r);
#pragma unroll
      for (int t = 0; t < 8; ++t) {
        int col = n0 + t * 16 + ln;
        float v = acc[t][r] + bias[col];
        outF[row * 1024 + col]  = v;
        out16[row * 1024 + col] = (_Float16)v;
      }
    }
  } else if (MODE == 1) {
    const int h = n0 >> 7;  // tile width 128 == one head
#pragma unroll
    for (int r = 0; r < 8; ++r) {
      float vals[8];
      float ss = 0.f;
#pragma unroll
      for (int t = 0; t < 8; ++t) {
        float v = acc[t][r] + bias[n0 + t * 16 + ln];
        vals[t] = v;
        ss += v * v;
      }
      ss += __shfl_xor(ss, 1, 32); ss += __shfl_xor(ss, 2, 32);
      ss += __shfl_xor(ss, 4, 32); ss += __shfl_xor(ss, 8, 32);
      float rinv = rsqrtf(ss * (1.f / 128.f) + 1e-6f);
      int row = rowbase + r, b = row >> 10, n = row & 1023;
      _Float16* dst = out16 + (((size_t)(b * 8 + h)) * 1024 + n) * 128;
#pragma unroll
      for (int t = 0; t < 8; ++t) {
        int c = t * 16 + ln;
        dst[c] = (_Float16)(vals[t] * rinv * nrmw[c]);
      }
    }
  } else if (MODE == 2) {
    const int h = n0 >> 7;
#pragma unroll
    for (int r = 0; r < 8; ++r) {
      int row = rowbase + r, b = row >> 10, n = row & 1023;
      _Float16* dst = out16 + ((size_t)(b * 8 + h) * 128) * 1024 + n;  // [bh][d][n]
#pragma unroll
      for (int t = 0; t < 8; ++t) {
        int d = t * 16 + ln;
        dst[(size_t)d * 1024] = (_Float16)(acc[t][r] + bias[n0 + d]);
      }
    }
  } else {  // MODE 3: + residual, fp32 pre-LayerNorm
#pragma unroll
    for (int r = 0; r < 8; ++r) {
      const size_t row = (size_t)(rowbase + r);
#pragma unroll
      for (int t = 0; t < 8; ++t) {
        int col = n0 + t * 16 + ln;
        float v = acc[t][r] + bias[col] + resid[row * 1024 + col];
        outF[row * 1024 + col] = v;
      }
    }
  }
}

// ---------------------------------------------------------------------------
// Flash attention, one workgroup per (128-query tile, b*H+h). 8 waves, each
// wave owns 16 query rows. Q fragments cached in registers. K blocks are
// async-staged into double-buffered LDS; V^T block single-buffered (its fill
// overlaps the S-compute + softmax phase). P goes through one LDS tile for
// the C->A layout conversion. ASYNCcnt in-order completion gives the waits:
// issue order ... V(j-1), K(j), V(j), K(j+1):  wait<=16 -> K(j) resident,
// wait<=8 -> V(j) resident.
// ---------------------------------------------------------------------------
__global__ void __launch_bounds__(256)
attn_k(const _Float16* __restrict__ q16, const _Float16* __restrict__ k16,
       const _Float16* __restrict__ vT16, _Float16* __restrict__ o16)
{
  __shared__ __align__(16) _Float16 Ks[2][128][136];  // key block, [n][d]
  __shared__ __align__(16) _Float16 Vs[128][136];     // V^T block, [d][key]
  __shared__ __align__(16) _Float16 Ps[128][136];     // probabilities, [m][key]

  const int tid  = threadIdx.x;
  const int w    = tid >> 5;
  const int lane = tid & 31;
  const int hi   = lane >> 4;
  const int ln   = lane & 15;
  const int bh   = blockIdx.y, b = bh >> 3, h = bh & 7;
  const int m0   = blockIdx.x * 128;

  const _Float16* qb  = q16  + ((size_t)bh * 1024 + m0) * 128;
  const _Float16* kb  = k16  + (size_t)bh * 1024 * 128;
  const _Float16* vtb = vT16 + (size_t)bh * 128 * 1024;   // [d][n]

  v16h qf[4];
#pragma unroll
  for (int ks = 0; ks < 4; ++ks) qf[ks] = ld_frag(qb, 128, w * 16, ks * 32);

  v8f Oacc[8];
#pragma unroll
  for (int t = 0; t < 8; ++t) Oacc[t] = 0.f;
  float mx[8], sm[8];
#pragma unroll
  for (int r = 0; r < 8; ++r) { mx[r] = -3.0e38f; sm[r] = 0.f; }
  const float scl = 0.08838834764831845f;  // Dh^-0.5, Dh=128

  // prefetch first K block
  async_tile_128((uint32_t)(uintptr_t)&Ks[0][0][0], kb, 128, tid);

  for (int j = 0; j < 8; ++j) {
    __syncthreads();  // everyone finished iteration j-1 (buffers reusable)
    // V^T block j (columns j*128..) -> Vs ; next K block -> other Ks buffer
    async_tile_128((uint32_t)(uintptr_t)&Vs[0][0], vtb + j * 128, 1024, tid);
    if (j < 7) {
      async_tile_128((uint32_t)(uintptr_t)&Ks[(j + 1) & 1][0][0],
                     kb + (size_t)(j + 1) * 128 * 128, 128, tid);
      asm volatile("s_wait_asynccnt 0x10" ::: "memory");  // K(j) resident
    } else {
      asm volatile("s_wait_asynccnt 0x8"  ::: "memory");  // K(7) resident
    }
    __syncthreads();  // K(j) visible to all waves

    // S = Q * K^T for key block j
    v8f Sac[8];
#pragma unroll
    for (int t = 0; t < 8; ++t) Sac[t] = 0.f;
#pragma unroll
    for (int ks = 0; ks < 4; ++ks) {
#pragma unroll
      for (int t = 0; t < 8; ++t) {
        v16h bf = ld_frag(&Ks[j & 1][0][0], 136, t * 16, ks * 32);
        Sac[t] = wmma32(qf[ks], bf, Sac[t]);
      }
    }

    // online softmax; each row's 128 scores live in 16 lanes x 8 tiles
#pragma unroll
    for (int r = 0; r < 8; ++r) {
      float vmx = -3.0e38f;
#pragma unroll
      for (int t = 0; t < 8; ++t) vmx = fmaxf(vmx, Sac[t][r]);
      vmx *= scl;
      vmx = fmaxf(vmx, __shfl_xor(vmx, 1, 32));
      vmx = fmaxf(vmx, __shfl_xor(vmx, 2, 32));
      vmx = fmaxf(vmx, __shfl_xor(vmx, 4, 32));
      vmx = fmaxf(vmx, __shfl_xor(vmx, 8, 32));
      float newm  = fmaxf(mx[r], vmx);
      float alpha = __expf(mx[r] - newm);
      mx[r] = newm;
      float rs = 0.f;
#pragma unroll
      for (int t = 0; t < 8; ++t) {
        float p = __expf(Sac[t][r] * scl - newm);
        rs += p;
        Ps[w * 16 + hi * 8 + r][t * 16 + ln] = (_Float16)p;
      }
      rs += __shfl_xor(rs, 1, 32); rs += __shfl_xor(rs, 2, 32);
      rs += __shfl_xor(rs, 4, 32); rs += __shfl_xor(rs, 8, 32);
      sm[r] = sm[r] * alpha + rs;
#pragma unroll
      for (int t = 0; t < 8; ++t) Oacc[t][r] *= alpha;
    }

    if (j < 7) asm volatile("s_wait_asynccnt 0x8" ::: "memory");  // V(j) ready
    else       asm volatile("s_wait_asynccnt 0x0" ::: "memory");
    __syncthreads();  // V(j) + Ps visible

    // O += P * V
#pragma unroll
    for (int ks = 0; ks < 4; ++ks) {
      v16h a = ld_frag(&Ps[0][0], 136, w * 16, ks * 32);
#pragma unroll
      for (int t = 0; t < 8; ++t) {
        v16h bf = ld_frag(&Vs[0][0], 136, t * 16, ks * 32);
        Oacc[t] = wmma32(a, bf, Oacc[t]);
      }
    }
  }

  // write O as f16 into the head-concatenated (B, N, D) layout
  _Float16* ob = o16 + ((size_t)b * 1024 + m0 + w * 16) * 1024 + h * 128;
#pragma unroll
  for (int r = 0; r < 8; ++r) {
    float inv = 1.0f / sm[r];
    int m = hi * 8 + r;
#pragma unroll
    for (int t = 0; t < 8; ++t)
      ob[(size_t)m * 1024 + t * 16 + ln] = (_Float16)(Oacc[t][r] * inv);
  }
}

// ---------------------------------------------------------------------------
__global__ void __launch_bounds__(256)
ln_k(const float* __restrict__ x, const float* __restrict__ gw,
     const float* __restrict__ gb, float* __restrict__ lf,
     _Float16* __restrict__ lf16)
{
  __shared__ float red[256];
  const int row = blockIdx.x, tid = threadIdx.x;
  const float* xr = x + (size_t)row * 1024;
  float4 v = *(const float4*)(xr + tid * 4);
  red[tid] = v.x + v.y + v.z + v.w;
  __syncthreads();
  for (int off = 128; off > 0; off >>= 1) {
    if (tid < off) red[tid] += red[tid + off];
    __syncthreads();
  }
  float mu = red[0] * (1.f / 1024.f);
  __syncthreads();
  float d0 = v.x - mu, d1 = v.y - mu, d2 = v.z - mu, d3 = v.w - mu;
  red[tid] = d0 * d0 + d1 * d1 + d2 * d2 + d3 * d3;
  __syncthreads();
  for (int off = 128; off > 0; off >>= 1) {
    if (tid < off) red[tid] += red[tid + off];
    __syncthreads();
  }
  float rstd = rsqrtf(red[0] * (1.f / 1024.f) + 1e-5f);
  float d[4] = {d0, d1, d2, d3};
#pragma unroll
  for (int i = 0; i < 4; ++i) {
    int c = tid * 4 + i;
    float y = d[i] * rstd * gw[c] + gb[c];
    lf[(size_t)row * 1024 + c]   = y;
    lf16[(size_t)row * 1024 + c] = (_Float16)y;
  }
}

__global__ void __launch_bounds__(256)
pool_k(const float* __restrict__ lf, float* __restrict__ pooled)
{
  int idx = blockIdx.x * 256 + threadIdx.x;  // b*1024 + d
  int b = idx >> 10, d = idx & 1023;
  const float* p = lf + ((size_t)b * 1024) * 1024 + d;
  float s = 0.f;
  for (int n = 0; n < 1024; ++n) s += p[(size_t)n * 1024];
  pooled[idx] = s * (1.f / 1024.f);
}

__global__ void __launch_bounds__(256)
out_k(const float* __restrict__ pooled, const float* __restrict__ W,
      const float* __restrict__ bias, float* __restrict__ out)
{
  int idx = blockIdx.x * 256 + threadIdx.x;  // b*512 + o
  int b = idx >> 9, o = idx & 511;
  const float* p = pooled + b * 1024;
  float s = bias[o];
  for (int d = 0; d < 1024; ++d) s += p[d] * W[(size_t)d * 512 + o];
  out[idx] = s;
}

// ---------------------------------------------------------------------------
extern "C" void kernel_launch(void* const* d_in, const int* in_sizes, int n_in,
                              void* d_out, int out_size, void* d_ws,
                              size_t ws_size, hipStream_t stream)
{
  const float* left  = (const float*)d_in[0];
  const float* right = (const float*)d_in[1];
  const float* Wsc   = (const float*)d_in[2];
  const float* bsc   = (const float*)d_in[3];
  const float* Wq    = (const float*)d_in[4];
  const float* bq    = (const float*)d_in[5];
  const float* Wk    = (const float*)d_in[6];
  const float* bk    = (const float*)d_in[7];
  const float* Wv    = (const float*)d_in[8];
  const float* bv    = (const float*)d_in[9];
  const float* Wo    = (const float*)d_in[10];
  const float* bo    = (const float*)d_in[11];
  const float* qnw   = (const float*)d_in[12];
  const float* knw   = (const float*)d_in[13];
  const float* lnw   = (const float*)d_in[14];
  const float* lnb   = (const float*)d_in[15];
  const float* Wout  = (const float*)d_in[16];
  const float* bout  = (const float*)d_in[17];
  float* out = (float*)d_out;

  char* ws = (char*)d_ws;
  size_t off = 0;
  auto take = [&](size_t bytes) {
    char* p = ws + off;
    off = (off + bytes + 255) & ~(size_t)255;
    return p;
  };
  const size_t BN = (size_t)16 * 1024;  // B*N rows
  float*    lf     = (float*)take(BN * 1024 * 4);
  float*    rf     = (float*)take(BN * 1024 * 4);
  _Float16* lf16   = (_Float16*)take(BN * 1024 * 2);
  _Float16* rf16   = (_Float16*)take(BN * 1024 * 2);
  _Float16* q16    = (_Float16*)take(BN * 1024 * 2);
  _Float16* k16    = (_Float16*)take(BN * 1024 * 2);
  _Float16* vT16   = (_Float16*)take(BN * 1024 * 2);
  _Float16* o16    = (_Float16*)take(BN * 1024 * 2);
  float*    xbuf   = (float*)take(BN * 1024 * 4);
  float*    pooled = (float*)take(16 * 1024 * 4);

  dim3 blk(256), g(8, 128);  // 8 n-tiles x 128 m-tiles of 128x128

  gemm_k<0><<<g, blk, 0, stream>>>(left,  nullptr, Wsc, bsc, nullptr, nullptr,
                                   lf, lf16, 3072);
  gemm_k<0><<<g, blk, 0, stream>>>(right, nullptr, Wsc, bsc, nullptr, nullptr,
                                   rf, rf16, 3072);
  for (int i = 0; i < 2; ++i) {
    const size_t wofs = (size_t)i * 1024 * 1024, bofs = (size_t)i * 1024;
    gemm_k<1><<<g, blk, 0, stream>>>(nullptr, lf16, Wq + wofs, bq + bofs,
                                     qnw + i * 128, nullptr, nullptr, q16, 1024);
    gemm_k<1><<<g, blk, 0, stream>>>(nullptr, rf16, Wk + wofs, bk + bofs,
                                     knw + i * 128, nullptr, nullptr, k16, 1024);
    gemm_k<2><<<g, blk, 0, stream>>>(nullptr, rf16, Wv + wofs, bv + bofs,
                                     nullptr, nullptr, nullptr, vT16, 1024);
    attn_k<<<g, blk, 0, stream>>>(q16, k16, vT16, o16);
    gemm_k<3><<<g, blk, 0, stream>>>(nullptr, o16, Wo + wofs, bo + bofs,
                                     nullptr, lf, xbuf, nullptr, 1024);
    ln_k<<<dim3(16384), blk, 0, stream>>>(xbuf, lnw + bofs, lnb + bofs, lf, lf16);
  }
  pool_k<<<dim3(64), blk, 0, stream>>>(lf, pooled);
  out_k<<<dim3(32), blk, 0, stream>>>(pooled, Wout, bout, out);

  (void)in_sizes; (void)n_in; (void)out_size; (void)ws_size;
}
